// Transformer_48223892799998
// MI455X (gfx1250) — matrix operand order repaired and checked by reading, not compiled, not run on hardware
//
#include <hip/hip_runtime.h>
#include <math.h>

#define N_NODES 16000
#define N_EDGES 256000
#define H_STR   72      // padded LDS row stride (bf16 elems) for hidden activations
#define PW_STR  264     // padded LDS row stride (f32 elems) for per-edge path weights

typedef __bf16 bf16_t;
typedef bf16_t v16bf __attribute__((ext_vector_type(16)));
typedef float  v8f   __attribute__((ext_vector_type(8)));

#define INV_SQ3 0.57735026918962576f

__device__ __forceinline__ v8f wmma_bf16(v16bf a, v16bf b, v8f c) {
  // D = A(16x32 bf16) x B(32x16 bf16) + C(16x16 f32)
  return __builtin_amdgcn_wmma_f32_16x16x32_bf16(false, a, false, b, (short)0, c, false, false);
}

__device__ __forceinline__ float gelu_tanh(float x) {
  const float A = 0.7978845608028654f, B = 0.044715f;
  return 0.5f * x * (1.0f + tanhf(A * (x + B * x * x * x)));
}

// A fragment (16xK slice, K-offset kofs) from global f32; row stride rs, col stride cs.
// Layout: lane -> M = lane&15; VGPR v holds K pair k=(v<4?0:16)+(lane>=16?8:0)+2*(v&3).
__device__ __forceinline__ v16bf load_a_g(const float* p, int rs, int cs, int lane, int kofs) {
  int m = lane & 15;
  int hi = (lane & 16) ? 8 : 0;
  const float* r = p + m * rs;
  union { bf16_t e[16]; v16bf v; } t;
#pragma unroll
  for (int vv = 0; vv < 8; ++vv) {
    int k = kofs + ((vv < 4) ? 0 : 16) + hi + 2 * (vv & 3);
    t.e[2 * vv]     = (bf16_t)r[k * cs];
    t.e[2 * vv + 1] = (bf16_t)r[(k + 1) * cs];
  }
  return t.v;
}

// A fragment from LDS bf16 (row-major, row stride `stride` elements), dword loads.
__device__ __forceinline__ v16bf load_a_lds(const bf16_t* p, int stride, int lane, int kofs) {
  int m = lane & 15;
  int hi = (lane & 16) ? 8 : 0;
  const unsigned int* q = (const unsigned int*)(p + m * stride + kofs + hi);
  union { unsigned int u[8]; v16bf v; } t;
#pragma unroll
  for (int vv = 0; vv < 8; ++vv)
    t.u[vv] = q[((vv < 4) ? 0 : 8) + (vv & 3)];
  return t.v;
}

// B fragment: this lane's column, two runs of 8 bf16 K-values, at base+8h and base+16+8h.
// (weights stored transposed in LDS: wt[col][K]); p = base + 16*(lane>=16); element s holds K = 8h+s (s<8), 16+8h+(s-8) (s>=8).
__device__ __forceinline__ v16bf load_b_lds(const bf16_t* p) {
  const unsigned int* q = (const unsigned int*)(p - ((__lane_id() >> 4) & 1) * 8);
  union { unsigned int u[8]; v16bf v; } t;
#pragma unroll
  for (int i = 0; i < 8; ++i) t.u[i] = q[((i < 4) ? 0 : 8) + (i & 3)];
  return t.v;
}

// Per-wave 16-edge MLP: emb(16x32) -> gelu -> (16x64) -> gelu -> pw(16x256) into pw_s.
__device__ void mlp_tile(const float* emb_rows,
                         const bf16_t* wt1, const bf16_t* wt2, const bf16_t* wtp,
                         bf16_t* h_s, float* pw_s, int lane)
{
  int nn = lane & 15;
  int hi16 = lane & 16;
  int mb = hi16 ? 8 : 0;

  // GEMM1: (16x32)@(32x64), scale 1/sqrt(32), gelu
  v16bf a1 = load_a_g(emb_rows, 32, 1, lane, 0);
  v8f acc[4];
#pragma unroll
  for (int j = 0; j < 4; ++j) {
    v8f c = {0.f,0.f,0.f,0.f,0.f,0.f,0.f,0.f};
    acc[j] = wmma_bf16(a1, load_b_lds(wt1 + (j*16 + nn)*32 + hi16), c);
  }
  const float s1 = 0.17677669529663687f;
#pragma unroll
  for (int j = 0; j < 4; ++j)
#pragma unroll
    for (int v = 0; v < 8; ++v)
      h_s[(mb + v)*H_STR + j*16 + nn] = (bf16_t)gelu_tanh(acc[j][v] * s1);
  asm volatile("" ::: "memory");   // same-wave LDS ops are in-order; block compile-time reordering

  // GEMM2: (16x64)@(64x64), scale 1/8, gelu
  v16bf alo = load_a_lds(h_s, H_STR, lane, 0);
  v16bf ahi = load_a_lds(h_s, H_STR, lane, 32);
#pragma unroll
  for (int j = 0; j < 4; ++j) {
    v8f c = {0.f,0.f,0.f,0.f,0.f,0.f,0.f,0.f};
    c = wmma_bf16(alo, load_b_lds(wt2 + (j*16 + nn)*64 + hi16), c);
    c = wmma_bf16(ahi, load_b_lds(wt2 + (j*16 + nn)*64 + 32 + hi16), c);
    acc[j] = c;
  }
  asm volatile("" ::: "memory");
#pragma unroll
  for (int j = 0; j < 4; ++j)
#pragma unroll
    for (int v = 0; v < 8; ++v)
      h_s[(mb + v)*H_STR + j*16 + nn] = (bf16_t)gelu_tanh(acc[j][v] * 0.125f);
  asm volatile("" ::: "memory");

  // GEMM3: (16x64)@(64x256), scale 1/8 -> pw_s (f32)
  alo = load_a_lds(h_s, H_STR, lane, 0);
  ahi = load_a_lds(h_s, H_STR, lane, 32);
  float* prow = pw_s + mb*PW_STR + nn;
#pragma unroll
  for (int j = 0; j < 16; ++j) {
    v8f c = {0.f,0.f,0.f,0.f,0.f,0.f,0.f,0.f};
    c = wmma_bf16(alo, load_b_lds(wtp + (j*16 + nn)*64 + hi16), c);
    c = wmma_bf16(ahi, load_b_lds(wtp + (j*16 + nn)*64 + 32 + hi16), c);
    float* pj = prow + j*16;
#pragma unroll
    for (int v = 0; v < 8; ++v)
      pj[v*PW_STR] = c[v] * 0.125f;
  }
  asm volatile("" ::: "memory");
}

// K1: per-node bilinear precompute  t_p[n, v, h] = sum_u x[n,u,(i)] * Wd[p][u,v,h]
// as (N x 64) @ (64 x 256) WMMA GEMMs.  Templated on <ACS, OSTRIDE> so the C-store
// offsets (v*OSTRIDE, j*16) fold into global_store_b32 24-bit immediates off one base.
template<int ACS, int OSTRIDE>
__global__ void __launch_bounds__(128) node_precompute_kernel(
    const float* __restrict__ node_f, const float* __restrict__ Wd,
    float* __restrict__ t0, float* __restrict__ t1,
    float* __restrict__ t2, float* __restrict__ t3)
{
  extern __shared__ char smem[];
  bf16_t* wt = (bf16_t*)smem;     // 256 cols x 64 K, transposed
  int p, aoff;
  float* outp;
  if (ACS == 1) {                 // scalar configs: blockIdx.y in {0,1}
    p = blockIdx.y; aoff = 0;
    outp = (p == 0) ? t0 : t1;
  } else {                        // vector configs: blockIdx.y in {0..5}
    int i = blockIdx.y % 3;
    p = 2 + blockIdx.y / 3; aoff = 64 + i;
    outp = ((p == 2) ? t2 : t3) + i*256;
  }
  const float* wdp = Wd + p * 16384;   // Wd[p] is (u=64, v*4+h=256) row-major
  for (int idx = threadIdx.x; idx < 256*64; idx += blockDim.x) {
    int col = idx >> 6, k = idx & 63;
    wt[idx] = (bf16_t)wdp[k*256 + col];
  }
  __syncthreads();
  int wave = threadIdx.x >> 5, lane = threadIdx.x & 31;
  int nodebase = (blockIdx.x*4 + wave) * 16;
  if (nodebase >= N_NODES) return;
  const float* arow = node_f + (size_t)nodebase*256 + aoff;
  v16bf alo = load_a_g(arow, 256, ACS, lane, 0);
  v16bf ahi = load_a_g(arow, 256, ACS, lane, 32);
  int nn = lane & 15, hi16 = lane & 16, mb = hi16 ? 8 : 0;
  // single 64-bit base per lane; all per-element offsets are compile-time constants
  float* obase = outp + (size_t)(nodebase + mb)*OSTRIDE + nn;
#pragma unroll
  for (int j = 0; j < 16; ++j) {
    v8f c = {0.f,0.f,0.f,0.f,0.f,0.f,0.f,0.f};
    c = wmma_bf16(alo, load_b_lds(wt + (j*16 + nn)*64 + hi16), c);
    c = wmma_bf16(ahi, load_b_lds(wt + (j*16 + nn)*64 + 32 + hi16), c);
#pragma unroll
    for (int v = 0; v < 8; ++v)
      obase[j*16 + v*OSTRIDE] = c[v];
  }
}

// Prefetch one 1KB-ish gather region (stride-32-float lines) into WGP caches.
__device__ __forceinline__ void prefetch_lines(const float* p, int nlines) {
  for (int l = 0; l < nlines; ++l)
    __builtin_prefetch(p + l*32, 0, 1);   // -> global_prefetch_b8 per 128B line
}

// K2: key-MLP (WMMA) + attention scores; expv to ws, z via atomics.
__global__ void __launch_bounds__(128) edge_score_kernel(
    const int* __restrict__ edge_src, const int* __restrict__ edge_dst,
    const float* __restrict__ esa, const float* __restrict__ eattr,
    const float* __restrict__ cutoff, const float* __restrict__ node_f,
    const float* __restrict__ Wk1, const float* __restrict__ Wk2, const float* __restrict__ Wkp,
    const float* __restrict__ t0, const float* __restrict__ t1,
    const float* __restrict__ t2, const float* __restrict__ t3,
    float* __restrict__ zacc, float* __restrict__ expv)
{
  extern __shared__ char smem[];
  bf16_t* wt1 = (bf16_t*)smem;          // 64 x 32
  bf16_t* wt2 = wt1 + 2048;             // 64 x 64
  bf16_t* wtp = wt2 + 4096;             // 256 x 64
  bf16_t* hs_all = wtp + 16384;
  float* pw_all = (float*)(smem + 45056 + (size_t)4*16*H_STR*2);
  int tid = threadIdx.x;
  for (int idx = tid; idx < 2048; idx += blockDim.x) { int n = idx >> 5, k = idx & 31; wt1[idx] = (bf16_t)Wk1[k*64 + n]; }
  for (int idx = tid; idx < 4096; idx += blockDim.x) { int n = idx >> 6, k = idx & 63; wt2[idx] = (bf16_t)Wk2[k*64 + n]; }
  for (int idx = tid; idx < 16384; idx += blockDim.x) {
    int col = idx >> 6, k = idx & 63; int pp = col >> 6, u = col & 63;
    wtp[idx] = (bf16_t)Wkp[(pp*64 + k)*64 + u];
  }
  __syncthreads();
  int wave = tid >> 5, lane = tid & 31;
  int tile = blockIdx.x*4 + wave;
  if (tile >= N_EDGES/16) return;
  bf16_t* h_s = hs_all + wave*16*H_STR;
  float*  pw_s = pw_all + wave*16*PW_STR;
  size_t ebase = (size_t)tile*16;

  // Resolve per-edge gather addresses up front and prefetch the L2-resident
  // t-tensors / node row so the fills overlap the 44 WMMAs of the MLP.
  int eL = lane & 15, uh = lane >> 4;
  size_t e = ebase + eL;
  int src = edge_src[e], dst = edge_dst[e];
  int ubase = uh*32;
  const float* nf_s = node_f + (size_t)src*256;
  const float* T0 = t0 + (size_t)dst*256 + ubase*4;
  const float* T1 = t1 + (size_t)dst*256 + ubase*4;
  const float* T2 = t2 + (size_t)dst*768 + ubase*4;
  const float* T3 = t3 + (size_t)dst*768 + ubase*4;
  prefetch_lines(T0, 4);
  prefetch_lines(T1, 4);
#pragma unroll
  for (int i = 0; i < 3; ++i) {
    prefetch_lines(T2 + i*256, 4);
    prefetch_lines(T3 + i*256, 4);
  }
  prefetch_lines(nf_s + ubase, 1);           // xs slice (128B)
  prefetch_lines(nf_s + 64 + ubase*3, 3);    // xv slice (384B)

  mlp_tile(esa + ebase*32, wt1, wt2, wtp, h_s, pw_s, lane);

  float e0  = eattr[e*4 + 0];
  float e1x = eattr[e*4 + 1], e1y = eattr[e*4 + 2], e1z = eattr[e*4 + 3];
  const float* pwrow = pw_s + eL*PW_STR;
  float acc[4] = {0.f,0.f,0.f,0.f};
  for (int uu = 0; uu < 32; ++uu) {
    int u = ubase + uu;
    float xs  = nf_s[u];
    float xv0 = nf_s[64 + u*3], xv1 = nf_s[64 + u*3 + 1], xv2 = nf_s[64 + u*3 + 2];
    float p0 = pwrow[u], p1 = pwrow[64 + u], p2 = pwrow[128 + u], p3 = pwrow[192 + u];
    float k0a = p0 * xs * e0;
    float dotv = xv0*e1x + xv1*e1y + xv2*e1z;
    float k0b = p3 * dotv * INV_SQ3;
    float c1a = p1 * xs * INV_SQ3;
    float c1b = p2 * e0 * INV_SQ3;
#pragma unroll
    for (int h = 0; h < 4; ++h) {
      float t0v = T0[uu*4 + h];
      float t1v = T1[uu*4 + h];
      float t2s = e1x*T2[uu*4 + h] + e1y*T2[256 + uu*4 + h] + e1z*T2[512 + uu*4 + h];
      float t3s = xv0*T3[uu*4 + h] + xv1*T3[256 + uu*4 + h] + xv2*T3[512 + uu*4 + h];
      acc[h] += k0a*t0v + k0b*t1v + c1a*t2s + c1b*t3s;
    }
  }
#pragma unroll
  for (int h = 0; h < 4; ++h) acc[h] += __shfl_xor(acc[h], 16, 32);
  if (uh == 0) {
    const float scale = 1.0f / 128.0f;   // 1/MUL/sqrt(4)
    float cut = cutoff[e];
#pragma unroll
    for (int h = 0; h < 4; ++h) {
      float ev = cut * expf(acc[h] * scale);
      expv[e*4 + h] = ev;
      atomicAdd(&zacc[(size_t)dst*4 + h], ev);
    }
  }
}

// K3: value-MLP (WMMA) + alpha weighting + scatter-add into node_out.
__global__ void __launch_bounds__(128) edge_value_kernel(
    const int* __restrict__ edge_src, const int* __restrict__ edge_dst,
    const float* __restrict__ esa, const float* __restrict__ eattr,
    const float* __restrict__ node_f,
    const float* __restrict__ Wv1, const float* __restrict__ Wv2, const float* __restrict__ Wvp,
    const float* __restrict__ zacc, const float* __restrict__ expv,
    float* __restrict__ nout)
{
  extern __shared__ char smem[];
  bf16_t* wt1 = (bf16_t*)smem;
  bf16_t* wt2 = wt1 + 2048;
  bf16_t* wtp = wt2 + 4096;
  bf16_t* hs_all = wtp + 16384;
  float* pw_all = (float*)(smem + 45056 + (size_t)4*16*H_STR*2);
  int tid = threadIdx.x;
  for (int idx = tid; idx < 2048; idx += blockDim.x) { int n = idx >> 5, k = idx & 31; wt1[idx] = (bf16_t)Wv1[k*64 + n]; }
  for (int idx = tid; idx < 4096; idx += blockDim.x) { int n = idx >> 6, k = idx & 63; wt2[idx] = (bf16_t)Wv2[k*64 + n]; }
  for (int idx = tid; idx < 16384; idx += blockDim.x) {
    int col = idx >> 6, k = idx & 63; int pp = col >> 6, u = col & 63;
    wtp[idx] = (bf16_t)Wvp[(pp*64 + k)*64 + u];
  }
  __syncthreads();
  int wave = tid >> 5, lane = tid & 31;
  int tile = blockIdx.x*4 + wave;
  if (tile >= N_EDGES/16) return;
  bf16_t* h_s = hs_all + wave*16*H_STR;
  float*  pw_s = pw_all + wave*16*PW_STR;
  size_t ebase = (size_t)tile*16;

  int eL = lane & 15, uh = lane >> 4;
  size_t e = ebase + eL;
  int src = edge_src[e], dst = edge_dst[e];
  int ubase = uh*32;
  const float* nf_s = node_f + (size_t)src*256;
  prefetch_lines(nf_s + ubase, 1);
  prefetch_lines(nf_s + 64 + ubase*3, 3);
  __builtin_prefetch(zacc + (size_t)dst*4, 0, 1);
  __builtin_prefetch(expv + e*4, 0, 1);

  mlp_tile(esa + ebase*32, wt1, wt2, wtp, h_s, pw_s, lane);

  float e0  = eattr[e*4 + 0];
  float e1x = eattr[e*4 + 1], e1y = eattr[e*4 + 2], e1z = eattr[e*4 + 3];
  float sa[4];
#pragma unroll
  for (int h = 0; h < 4; ++h) {
    float zz = zacc[(size_t)dst*4 + h];
    zz = (zz == 0.0f) ? 1.0f : zz;
    float a = expv[e*4 + h] / zz;
    sa[h] = sqrtf(fmaxf(a, 0.0f));
  }
  const float* pwrow = pw_s + eL*PW_STR;
  float* NO = nout + (size_t)dst*512;
#pragma unroll 1
  for (int half = 0; half < 2; ++half) {
    float s = sa[uh*2 + half];                 // head = u>>4, loop-invariant per half
    for (int uu = half*16; uu < half*16 + 16; ++uu) {
      int u = ubase + uu;
      float xs  = nf_s[u];
      float xv0 = nf_s[64+u*3], xv1 = nf_s[64+u*3+1], xv2 = nf_s[64+u*3+2];
      float p0 = pwrow[u], p1 = pwrow[64+u], p2 = pwrow[128+u], p3 = pwrow[192+u];
      atomicAdd(&NO[u], s * p0 * xs * e0);
      float dotv = xv0*e1x + xv1*e1y + xv2*e1z;
      atomicAdd(&NO[64 + u], s * p3 * dotv * INV_SQ3);
      float c1 = s * p1 * xs;
      atomicAdd(&NO[128 + u*3 + 0], c1 * e1x);
      atomicAdd(&NO[128 + u*3 + 1], c1 * e1y);
      atomicAdd(&NO[128 + u*3 + 2], c1 * e1z);
      float c2 = s * p2 * e0;
      atomicAdd(&NO[320 + u*3 + 0], c2 * xv0);
      atomicAdd(&NO[320 + u*3 + 1], c2 * xv1);
      atomicAdd(&NO[320 + u*3 + 2], c2 * xv2);
    }
  }
}

// K4: per-node output linears (small: 0.5 GMAC, L2-resident)
__global__ void __launch_bounds__(256) final_linear_kernel(
    const float* __restrict__ nout, const float* __restrict__ WL0,
    const float* __restrict__ WL1, float* __restrict__ out)
{
  int gid = blockIdx.x*256 + threadIdx.x;
  if (gid >= N_NODES*256) return;
  int n = gid >> 8, c = gid & 255;
  const float* no = nout + (size_t)n*512;
  const float ls = 0.08838834764831845f;  // 1/sqrt(128)
  float acc = 0.f;
  if (c < 64) {
    for (int u = 0; u < 128; ++u) acc += no[u] * WL0[u*64 + c];
  } else {
    int cc = c - 64; int vo = cc / 3, i = cc - vo*3;
    for (int u = 0; u < 128; ++u) acc += no[128 + u*3 + i] * WL1[u*64 + vo];
  }
  out[gid] = acc * ls;
}

extern "C" void kernel_launch(void* const* d_in, const int* in_sizes, int n_in,
                              void* d_out, int out_size, void* d_ws, size_t ws_size,
                              hipStream_t stream)
{
  (void)in_sizes; (void)n_in; (void)out_size; (void)ws_size;
  const int*   edge_src = (const int*)d_in[0];
  const int*   edge_dst = (const int*)d_in[1];
  const float* esa      = (const float*)d_in[2];
  const float* eattr    = (const float*)d_in[3];
  const float* cutoff   = (const float*)d_in[4];
  const float* node_f   = (const float*)d_in[5];
  const float* Wk1 = (const float*)d_in[6];
  const float* Wk2 = (const float*)d_in[7];
  const float* Wkp = (const float*)d_in[8];
  const float* Wv1 = (const float*)d_in[9];
  const float* Wv2 = (const float*)d_in[10];
  const float* Wvp = (const float*)d_in[11];
  const float* Wd  = (const float*)d_in[12];
  const float* WL0 = (const float*)d_in[13];
  const float* WL1 = (const float*)d_in[14];
  float* out = (float*)d_out;

  float* ws = (float*)d_ws;
  float* t0  = ws;
  float* t1  = t0 + (size_t)N_NODES*256;
  float* t2  = t1 + (size_t)N_NODES*256;
  float* t3  = t2 + (size_t)N_NODES*768;
  float* zb  = t3 + (size_t)N_NODES*768;
  float* ev  = zb + (size_t)N_NODES*4;
  float* nout = ev + (size_t)N_EDGES*4;

  hipMemsetAsync(zb,   0, (size_t)N_NODES*4*sizeof(float),   stream);
  hipMemsetAsync(nout, 0, (size_t)N_NODES*512*sizeof(float), stream);

  const size_t smem_node = 256*64*sizeof(bf16_t);                           // 32 KB
  const size_t smem_edge = 45056 + (size_t)4*16*H_STR*2 + (size_t)4*16*PW_STR*4; // ~119 KB
  (void)hipFuncSetAttribute(reinterpret_cast<const void*>(node_precompute_kernel<1,256>),
                            hipFuncAttributeMaxDynamicSharedMemorySize, (int)smem_node);
  (void)hipFuncSetAttribute(reinterpret_cast<const void*>(node_precompute_kernel<3,768>),
                            hipFuncAttributeMaxDynamicSharedMemorySize, (int)smem_node);
  (void)hipFuncSetAttribute(reinterpret_cast<const void*>(edge_score_kernel),
                            hipFuncAttributeMaxDynamicSharedMemorySize, (int)smem_edge);
  (void)hipFuncSetAttribute(reinterpret_cast<const void*>(edge_value_kernel),
                            hipFuncAttributeMaxDynamicSharedMemorySize, (int)smem_edge);

  dim3 g1s(N_NODES/64, 2);   // t0, t1 (A = xs, contiguous)
  dim3 g1v(N_NODES/64, 6);   // t2, t3 x {i=0,1,2} (A = xv_i, stride 3)
  node_precompute_kernel<1,256><<<g1s, 128, smem_node, stream>>>(node_f, Wd, t0, t1, t2, t3);
  node_precompute_kernel<3,768><<<g1v, 128, smem_node, stream>>>(node_f, Wd, t0, t1, t2, t3);
  edge_score_kernel<<<N_EDGES/64, 128, smem_edge, stream>>>(
      edge_src, edge_dst, esa, eattr, cutoff, node_f, Wk1, Wk2, Wkp,
      t0, t1, t2, t3, zb, ev);
  edge_value_kernel<<<N_EDGES/64, 128, smem_edge, stream>>>(
      edge_src, edge_dst, esa, eattr, node_f, Wv1, Wv2, Wvp, zb, ev, nout);
  final_linear_kernel<<<(N_NODES*256)/256, 256, 0, stream>>>(nout, WL0, WL1, out);
}
